// GraphConvolution_3083786519088
// MI455X (gfx1250) — compile-verified, matrix operand order checked by reading, not run on hardware
//
#include <hip/hip_runtime.h>
#include <hip/hip_bf16.h>
#include <stddef.h>

// GCN layer: out = (D A D @ text) @ W + bias,  B=4, N=4096, F=300
// bf16 WMMA (v_wmma_f32_16x16x32_bf16), fp32 accumulate.
// D_j folded into text up-front; D_i applied in GEMM-1 epilogue.
// bf16 tiles staged with GLOBAL_LOAD_ASYNC_TO_LDS_B128 (ASYNCcnt);
// adj tile goes through VGPRs for the fp32->bf16 convert.

#define BATCH 4
#define NN    4096
#define FIN   300
#define FPAD  320          // padded feature dim (multiple of 64)

#define BM    128
#define BN    64
#define BK    32
#define ASTR  40           // LDS row stride in bf16 (pad vs 64-bank conflicts)

typedef __attribute__((ext_vector_type(4)))  __bf16 v4bf;
typedef __attribute__((ext_vector_type(8)))  __bf16 v8bf;
typedef __attribute__((ext_vector_type(16))) __bf16 v16bf;
typedef __attribute__((ext_vector_type(8)))  float  v8f;

typedef __attribute__((address_space(3))) char lds_char;

// ---------------------------------------------------------------------------
// Async 16-byte copy global -> LDS (CDNA5 GLOBAL_LOAD_ASYNC_TO_LDS_B128,
// tracked by ASYNCcnt).  VDST VGPR carries the LDS byte offset (ISA 15.18.3).
// ---------------------------------------------------------------------------
__device__ __forceinline__ void async_ld16(void* lds, const void* g) {
  lds_char* l = (lds_char*)lds;                 // addrspace(3): 32-bit offset
  asm volatile("global_load_async_to_lds_b128 %0, %1, off"
               :: "v"(l), "v"(g) : "memory");
}

__device__ __forceinline__ void wait_async0() {
#if __has_builtin(__builtin_amdgcn_s_wait_asynccnt)
  __builtin_amdgcn_s_wait_asynccnt(0);
#else
  asm volatile("s_wait_asynccnt 0x0" ::: "memory");
#endif
}

// ---------------------------------------------------------------------------
// Fragment load (16x32 bf16 operand) from an LDS tile with row stride ASTR.
// Per ISA 7.12.2: lane l<16 -> row=row0+l, K = {0..7, 16..23} (hi lanes +8).
// ---------------------------------------------------------------------------
__device__ __forceinline__ v16bf frag16x32(const __bf16* base, int row0) {
  const int lane = threadIdx.x & 31;
  const int r  = row0 + (lane & 15);
  const int kb = (lane >> 4) << 3;              // 0 or 8
  const __bf16* p = base + r * ASTR + kb;
  v8bf lo = *(const v8bf*)(p);                  // K = kb .. kb+7
  v8bf hi = *(const v8bf*)(p + 16);             // K = kb+16 .. kb+23
  v16bf f;
#pragma unroll
  for (int i = 0; i < 8; ++i) { f[i] = lo[i]; f[i + 8] = hi[i]; }
  return f;
}

__device__ __forceinline__ v8f wmma_bf16(v16bf a, v16bf b, v8f c) {
  return __builtin_amdgcn_wmma_f32_16x16x32_bf16(
      false, a, false, b, (short)0, c, false, false);
}

// ---------------------------------------------------------------------------
// Kernel 1: d[b,i] = rsqrt(rowsum(adj[b,i,:]) + 1).  One wave32 per row.
// ---------------------------------------------------------------------------
__global__ __launch_bounds__(256) void k_rowsum(const float* __restrict__ adj,
                                                float* __restrict__ dvec) {
  const int row  = blockIdx.x * 8 + (threadIdx.x >> 5);
  const int lane = threadIdx.x & 31;
  const float4* p4 = (const float4*)(adj + (size_t)row * NN);
  float s = 0.0f;
#pragma unroll 4
  for (int t = 0; t < NN / 128; ++t) {
    float4 v = p4[lane + t * 32];
    s += v.x + v.y + v.z + v.w;
  }
#pragma unroll
  for (int off = 16; off > 0; off >>= 1) s += __shfl_xor(s, off, 32);
  if (lane == 0) dvec[row] = rsqrtf(s + 1.0f);
}

// ---------------------------------------------------------------------------
// Kernel 2: textT[b,f,j] = bf16(d[b,j] * text[b,j,f]); zero-pad f in [300,320)
// ---------------------------------------------------------------------------
__global__ __launch_bounds__(256) void k_scale_text(const float* __restrict__ text,
                                                    const float* __restrict__ dvec,
                                                    __bf16* __restrict__ textT) {
  const int b    = blockIdx.y;
  const int j    = blockIdx.x * 8 + (threadIdx.x >> 5);
  const int lane = threadIdx.x & 31;
  const float d  = dvec[b * NN + j];
  const float* src = text + ((size_t)b * NN + j) * FIN;
  __bf16* dst = textT + (size_t)b * FPAD * NN + j;
  for (int f = lane; f < FPAD; f += 32) {
    float v = (f < FIN) ? d * src[f] : 0.0f;
    dst[(size_t)f * NN] = (__bf16)v;
  }
}

// ---------------------------------------------------------------------------
// Kernel 3: weightT[n,k] = bf16(weight[k,n]) zero-padded to 320x320.
// ---------------------------------------------------------------------------
__global__ __launch_bounds__(256) void k_pack_weight(const float* __restrict__ w,
                                                     __bf16* __restrict__ wT) {
  const int idx = blockIdx.x * 256 + threadIdx.x;
  if (idx >= FPAD * FPAD) return;
  const int n = idx / FPAD, k = idx % FPAD;
  float v = (n < FIN && k < FIN) ? w[k * FIN + n] : 0.0f;
  wT[idx] = (__bf16)v;
}

// ---------------------------------------------------------------------------
// Kernel 4: agg[b,i,f] = d[b,i] * (adj[b] @ textT[b]^T)  (bf16 out, padded F)
// Pipeline per K-step: issue A-loads (regs) + async B tile -> WMMAs ->
// commit A to LDS -> s_wait_asynccnt -> barrier.
// ---------------------------------------------------------------------------
__global__ __launch_bounds__(256) void k_agg_gemm(const float* __restrict__ adj,
                                                  const __bf16* __restrict__ textT,
                                                  const float* __restrict__ dvec,
                                                  __bf16* __restrict__ agg) {
  const int b  = blockIdx.z;
  const int m0 = blockIdx.y * BM;
  const int n0 = blockIdx.x * BN;

  const float*  A  = adj   + (size_t)b * NN * NN;     // [NN][NN] fp32
  const __bf16* Bt = textT + (size_t)b * FPAD * NN;   // [FPAD][NN] bf16

  __shared__ __align__(16) __bf16 sA[2][BM][ASTR];
  __shared__ __align__(16) __bf16 sB[2][BN][ASTR];

  const int tid  = threadIdx.x;
  const int lane = tid & 31;
  const int wid  = tid >> 5;
  const int wm   = (wid & 3) * 32;
  const int wn   = (wid >> 2) * 32;

  v8f acc[2][2];
#pragma unroll
  for (int tm = 0; tm < 2; ++tm)
#pragma unroll
    for (int tn = 0; tn < 2; ++tn)
#pragma unroll
      for (int e = 0; e < 8; ++e) acc[tm][tn][e] = 0.0f;

  float4 ra[4];                                  // staged A (128x32 fp32)

  auto issueA = [&](int kt) {                    // global loads -> regs
    const int k0 = kt * BK;
#pragma unroll
    for (int i = 0; i < 4; ++i) {
      const int idx = tid + i * 256;
      const int r = idx >> 3;
      const int c = (idx & 7) * 4;
      ra[i] = *(const float4*)(A + (size_t)(m0 + r) * NN + k0 + c);
      if (i == 0)  // speculative prefetch two K-steps ahead (OOB is dropped)
        __builtin_prefetch((const void*)(A + (size_t)(m0 + r) * NN + k0 + 2 * BK + c), 0, 0);
    }
  };
  auto commitA = [&](int buf) {                  // convert + LDS store
#pragma unroll
    for (int i = 0; i < 4; ++i) {
      const int idx = tid + i * 256;
      const int r = idx >> 3;
      const int c = (idx & 7) * 4;
      v4bf w;
      w[0] = (__bf16)ra[i].x; w[1] = (__bf16)ra[i].y;
      w[2] = (__bf16)ra[i].z; w[3] = (__bf16)ra[i].w;
      *(v4bf*)&sA[buf][r][c] = w;
    }
  };
  auto asyncB = [&](int buf, int kt) {           // DMA bf16 tile -> LDS
    const int k0 = kt * BK;
    const int r = tid >> 2;
    const int c = (tid & 3) * 8;
    async_ld16(&sB[buf][r][c], Bt + (size_t)(n0 + r) * NN + k0 + c);
  };

  issueA(0);
  asyncB(0, 0);
  commitA(0);
  wait_async0();
  __syncthreads();

  const int NK = NN / BK;   // 128
  for (int kt = 0; kt < NK; ++kt) {
    const int buf = kt & 1;
    if (kt + 1 < NK) { issueA(kt + 1); asyncB(buf ^ 1, kt + 1); }

    v16bf a0 = frag16x32(&sA[buf][0][0], wm);
    v16bf a1 = frag16x32(&sA[buf][0][0], wm + 16);
    v16bf b0 = frag16x32(&sB[buf][0][0], wn);
    v16bf b1 = frag16x32(&sB[buf][0][0], wn + 16);

    acc[0][0] = wmma_bf16(a0, b0, acc[0][0]);
    acc[0][1] = wmma_bf16(a0, b1, acc[0][1]);
    acc[1][0] = wmma_bf16(a1, b0, acc[1][0]);
    acc[1][1] = wmma_bf16(a1, b1, acc[1][1]);

    if (kt + 1 < NK) commitA(buf ^ 1);
    wait_async0();
    __syncthreads();
  }

  // Epilogue: scale by d_i, write bf16 agg.  C layout (7.12.2).
  const int cbase = lane & 15;
  const int rsel  = (lane >> 4) * 8;
#pragma unroll
  for (int tm = 0; tm < 2; ++tm)
#pragma unroll
    for (int tn = 0; tn < 2; ++tn) {
      const int col  = n0 + wn + tn * 16 + cbase;
      const int rowb = m0 + wm + tm * 16 + rsel;
#pragma unroll
      for (int r = 0; r < 8; ++r) {
        const int row = rowb + r;
        const float val = acc[tm][tn][r] * dvec[b * NN + row];
        agg[((size_t)b * NN + row) * FPAD + col] = (__bf16)val;
      }
    }
}

// ---------------------------------------------------------------------------
// Kernel 5: out[m, n] = agg[m, :] @ weightT[n, :] + bias[n]   (fp32 out)
// Both tiles are bf16 in memory -> both staged with async DMA to LDS.
// ---------------------------------------------------------------------------
__global__ __launch_bounds__(256) void k_out_gemm(const __bf16* __restrict__ agg,
                                                  const __bf16* __restrict__ weightT,
                                                  const float* __restrict__ bias,
                                                  float* __restrict__ out) {
  const int m0 = blockIdx.y * BM;   // gridDim.y = 128
  const int n0 = blockIdx.x * BN;   // gridDim.x = 5

  __shared__ __align__(16) __bf16 sA[2][BM][ASTR];
  __shared__ __align__(16) __bf16 sB[2][BN][ASTR];

  const int tid  = threadIdx.x;
  const int lane = tid & 31;
  const int wid  = tid >> 5;
  const int wm   = (wid & 3) * 32;
  const int wn   = (wid >> 2) * 32;

  v8f acc[2][2];
#pragma unroll
  for (int tm = 0; tm < 2; ++tm)
#pragma unroll
    for (int tn = 0; tn < 2; ++tn)
#pragma unroll
      for (int e = 0; e < 8; ++e) acc[tm][tn][e] = 0.0f;

  auto stage = [&](int buf, int kt) {            // all-async staging
    const int k0 = kt * BK;
#pragma unroll
    for (int i = 0; i < 2; ++i) {
      const int idx = tid + i * 256;
      const int r = idx >> 2;
      const int c = (idx & 3) * 8;
      async_ld16(&sA[buf][r][c], agg + (size_t)(m0 + r) * FPAD + k0 + c);
    }
    {
      const int r = tid >> 2;
      const int c = (tid & 3) * 8;
      async_ld16(&sB[buf][r][c], weightT + (size_t)(n0 + r) * FPAD + k0 + c);
    }
  };

  stage(0, 0);
  wait_async0();
  __syncthreads();

  const int NK = FPAD / BK;   // 10
  for (int kt = 0; kt < NK; ++kt) {
    const int buf = kt & 1;
    if (kt + 1 < NK) stage(buf ^ 1, kt + 1);

    v16bf a0 = frag16x32(&sA[buf][0][0], wm);
    v16bf a1 = frag16x32(&sA[buf][0][0], wm + 16);
    v16bf b0 = frag16x32(&sB[buf][0][0], wn);
    v16bf b1 = frag16x32(&sB[buf][0][0], wn + 16);

    acc[0][0] = wmma_bf16(a0, b0, acc[0][0]);
    acc[0][1] = wmma_bf16(a0, b1, acc[0][1]);
    acc[1][0] = wmma_bf16(a1, b0, acc[1][0]);
    acc[1][1] = wmma_bf16(a1, b1, acc[1][1]);

    wait_async0();
    __syncthreads();
  }

  const int cbase = lane & 15;
  const int rsel  = (lane >> 4) * 8;
#pragma unroll
  for (int tm = 0; tm < 2; ++tm)
#pragma unroll
    for (int tn = 0; tn < 2; ++tn) {
      const int col = n0 + wn + tn * 16 + cbase;
      if (col < FIN) {
        const float bv   = bias[col];
        const int   rowb = m0 + wm + tm * 16 + rsel;
#pragma unroll
        for (int r = 0; r < 8; ++r) {
          const int row = rowb + r;   // row = b*NN + i
          out[(size_t)row * FIN + col] = acc[tm][tn][r] + bv;
        }
      }
    }
}

// ---------------------------------------------------------------------------
extern "C" void kernel_launch(void* const* d_in, const int* in_sizes, int n_in,
                              void* d_out, int out_size, void* d_ws, size_t ws_size,
                              hipStream_t stream) {
  const float* text   = (const float*)d_in[0];   // [B, N, 300]
  const float* adj    = (const float*)d_in[1];   // [B, N, N]
  const float* weight = (const float*)d_in[2];   // [300, 300]
  const float* bias   = (const float*)d_in[3];   // [300]
  float* out          = (float*)d_out;           // [B, N, 300]

  char* ws = (char*)d_ws;
  float*  dvec    = (float*)(ws);                                   //  64 KB
  __bf16* textT   = (__bf16*)(ws + 65536);                          // 10.0 MB
  __bf16* weightT = (__bf16*)(ws + 65536 + 10485760);               // 200 KB
  __bf16* agg     = (__bf16*)(ws + 65536 + 10485760 + 204800);      // 10.0 MB

  (void)in_sizes; (void)n_in; (void)out_size; (void)ws_size;

  k_rowsum<<<dim3(BATCH * NN / 8), dim3(256), 0, stream>>>(adj, dvec);
  k_scale_text<<<dim3(NN / 8, BATCH), dim3(256), 0, stream>>>(text, dvec, textT);
  k_pack_weight<<<dim3((FPAD * FPAD + 255) / 256), dim3(256), 0, stream>>>(weight, weightT);
  k_agg_gemm<<<dim3(FPAD / BN, NN / BM, BATCH), dim3(256), 0, stream>>>(adj, textT, dvec, agg);
  k_out_gemm<<<dim3(FPAD / BN, (BATCH * NN) / BM), dim3(256), 0, stream>>>(agg, weightT, bias, out);
}
